// Mamba_32427003084810
// MI455X (gfx1250) — matrix operand order repaired and checked by reading, not compiled, not run on hardware
//
#include <hip/hip_runtime.h>

#define D_MODEL 768
#define D_INNER 1536
#define N_STATE 16
#define DT_RANK 48
#define D_CONV  4
#define N_LAYER 2
#define VOCAB   32000
#define BB      2
#define LL      2048
#define ROWS    (BB*LL)                 // 4096
#define XDBL_W  (DT_RANK + 2*N_STATE)   // 80
#define XPW_NPAD 256                    // x_proj N=80 padded to one BN tile

typedef __attribute__((ext_vector_type(16))) __bf16 v16bf;
typedef __attribute__((ext_vector_type(8)))  float  v8f;

__device__ __forceinline__ __bf16 f2bf(float f) {
  unsigned u = __builtin_bit_cast(unsigned, f);
  u += 0x7fffu + ((u >> 16) & 1u);          // round-to-nearest-even
  unsigned short s = (unsigned short)(u >> 16);
  return __builtin_bit_cast(__bf16, s);
}

__device__ __forceinline__ unsigned lds_off(const void* p) {
  // generic LDS pointer: low 32 bits are the workgroup-relative LDS byte offset
  return (unsigned)(unsigned long long)p;
}

// async 16-byte global -> LDS copy, tracked by ASYNCcnt
__device__ __forceinline__ void async_cp16(unsigned lds, const void* g) {
  asm volatile("global_load_async_to_lds_b128 %0, %1, off"
               :: "v"(lds), "v"(g) : "memory");
}
__device__ __forceinline__ void wait_async0() {
  asm volatile("s_wait_asynccnt 0x0" ::: "memory");
}
// allow the 5 just-issued (next-buffer) copies to remain in flight
__device__ __forceinline__ void wait_async5() {
  asm volatile("s_wait_asynccnt 0x5" ::: "memory");
}

// ================================================================ pre-pass
// fp32 -> bf16 elementwise (embedding is already [N][K] = [vocab][dmodel])
__global__ void k_cvt_bf16(const float* __restrict__ src, __bf16* __restrict__ dst, long n) {
  long i = (long)blockIdx.x * blockDim.x + threadIdx.x;
  if (i < n) dst[i] = f2bf(src[i]);
}

// fp32 [K][N] -> bf16 [N][K]  (K-contiguous "B^T" layout for the GEMM)
__global__ void k_w_transpose(const float* __restrict__ src, __bf16* __restrict__ dst,
                              int K, int N) {
  long i = (long)blockIdx.x * blockDim.x + threadIdx.x;
  if (i >= (long)N * K) return;
  int n = (int)(i / K), k = (int)(i - (long)n * K);
  dst[i] = f2bf(src[(long)k * N + n]);
}

// ================================================================ embedding
__global__ void k_embed(const int* __restrict__ ids, const float* __restrict__ emb,
                        float* __restrict__ x) {
  int i = blockIdx.x * blockDim.x + threadIdx.x;
  if (i >= ROWS * D_MODEL) return;
  int row = i / D_MODEL, c = i - row * D_MODEL;
  x[i] = emb[(long)ids[row] * D_MODEL + c];
}

// ================================================================ rmsnorm -> bf16
__global__ __launch_bounds__(256) void k_rmsnorm(const float* __restrict__ x,
                                                 const float* __restrict__ w,
                                                 __bf16* __restrict__ out) {
  __shared__ float red[256];
  int row = blockIdx.x;
  const float* xr = x + (long)row * D_MODEL;
  float s = 0.f;
  for (int c = threadIdx.x; c < D_MODEL; c += 256) { float v = xr[c]; s += v * v; }
  red[threadIdx.x] = s;
  __syncthreads();
  for (int st = 128; st > 0; st >>= 1) {
    if (threadIdx.x < st) red[threadIdx.x] += red[threadIdx.x + st];
    __syncthreads();
  }
  float scale = rsqrtf(1e-5f + red[0] * (1.0f / D_MODEL));
  for (int c = threadIdx.x; c < D_MODEL; c += 256)
    out[(long)row * D_MODEL + c] = f2bf(xr[c] * scale * w[c]);
}

// ================================================================ WMMA GEMM (bf16 in, f32 out)
// C[M,N] (+)= A[M,K] * B[K,N], A bf16 row-major [M][K], Bt bf16 [N][K].
// Block tile 64x256, 8 waves (2x4), wave tile 32x64 = 2x4 v_wmma_f32_16x16x32_bf16.
// Double-buffered LDS; tiles staged with global_load_async_to_lds_b128 and the
// next tile's copies overlapped with compute (s_wait_asynccnt 5).
#define BM 64
#define BN 256
#define BK 32
#define BKE 40    // LDS row stride in bf16 (32 data + 8 pad): 80B rows, 16B aligned

struct Q2 { uint4 a, b; };

__device__ __forceinline__ v16bf frag_a(const __bf16* rowp, int hi) {
  Q2 q;
  q.a = *(const uint4*)(rowp + hi * 8);        // K = hi*8 .. hi*8+7
  q.b = *(const uint4*)(rowp + 16 + hi * 8);   // K = 16+hi*8 ..
  return __builtin_bit_cast(v16bf, q);
}
__device__ __forceinline__ v16bf frag_b(const __bf16* rowp, int hi) {
  Q2 q;
  q.a = *(const uint4*)(rowp + hi * 8);        // K = hi*8 .. hi*8+7
  q.b = *(const uint4*)(rowp + 16 + hi * 8);   // K = 16+hi*8 ..
  return __builtin_bit_cast(v16bf, q);
}

template<bool ACC>
__global__ __launch_bounds__(256)
void k_gemm_bf16(const __bf16* __restrict__ A, const __bf16* __restrict__ Bt,
                 float* __restrict__ C, int N, int K, int ldc) {
  __shared__ __bf16 sA[2][BM * BKE];   // 2 x  5120 B
  __shared__ __bf16 sB[2][BN * BKE];   // 2 x 20480 B
  const int tid  = threadIdx.x;
  const int lane = tid & 31, wave = tid >> 5;
  const int wm = wave >> 2, wn = wave & 3;      // 2 x 4 wave grid
  const int lo = lane & 15, hi = lane >> 4;
  const int m0 = blockIdx.y * BM, n0 = blockIdx.x * BN;

  // per-thread 16B chunk assignment (4 chunks per 64B tile row)
  const int am = tid >> 2, ac = tid & 3;        // A: 256 chunks, 1/thread

  auto stage = [&](int buf, int kk) {
    async_cp16(lds_off(&sA[buf][am * BKE + ac * 8]),
               A + (long)(m0 + am) * K + kk + ac * 8);
    #pragma unroll
    for (int t = 0; t < 4; ++t) {               // B: 1024 chunks, 4/thread
      int chunk = tid + t * 256;
      int n = chunk >> 2, c = chunk & 3;
      async_cp16(lds_off(&sB[buf][n * BKE + c * 8]),
                 Bt + (long)(n0 + n) * K + kk + c * 8);
    }
  };

  v8f acc[2][4];
  v8f z = {0.f,0.f,0.f,0.f,0.f,0.f,0.f,0.f};
  #pragma unroll
  for (int i = 0; i < 2; ++i)
    #pragma unroll
    for (int j = 0; j < 4; ++j) acc[i][j] = z;

  const int nIter = K / BK;
  stage(0, 0);
  for (int it = 0; it < nIter; ++it) {
    const int cur = it & 1;
    if (it + 1 < nIter) { stage(1 - cur, (it + 1) * BK); wait_async5(); }
    else                { wait_async0(); }
    __syncthreads();

    const __bf16* sAc = sA[cur];
    const __bf16* sBc = sB[cur];
    v16bf a0 = frag_a(sAc + (wm * 32 + lo) * BKE, hi);
    v16bf a1 = frag_a(sAc + (wm * 32 + 16 + lo) * BKE, hi);
    v16bf bf[4];
    #pragma unroll
    for (int j = 0; j < 4; ++j)
      bf[j] = frag_b(sBc + (wn * 64 + j * 16 + lo) * BKE, hi);

    #pragma unroll
    for (int j = 0; j < 4; ++j) {
      acc[0][j] = __builtin_amdgcn_wmma_f32_16x16x32_bf16(false, a0, false, bf[j], (short)0, acc[0][j], false, false);
      acc[1][j] = __builtin_amdgcn_wmma_f32_16x16x32_bf16(false, a1, false, bf[j], (short)0, acc[1][j], false, false);
    }
    __syncthreads();
  }

  // epilogue: f32 C/D layout -> lane lo = N col, VGPR v + 8*hi = M row
  #pragma unroll
  for (int i = 0; i < 2; ++i) {
    #pragma unroll
    for (int j = 0; j < 4; ++j) {
      int c = n0 + wn * 64 + j * 16 + lo;
      if (c < N) {
        #pragma unroll
        for (int v = 0; v < 8; ++v) {
          int r = m0 + wm * 32 + i * 16 + hi * 8 + v;
          if (ACC) C[(long)r * ldc + c] += acc[i][j][v];
          else     C[(long)r * ldc + c]  = acc[i][j][v];
        }
      }
    }
  }
}

// ================================================================ depthwise conv + SiLU
__global__ void k_conv_silu(const float* __restrict__ xr, const float* __restrict__ cw,
                            const float* __restrict__ cb, float* __restrict__ xi,
                            __bf16* __restrict__ xib) {
  int i = blockIdx.x * blockDim.x + threadIdx.x;
  if (i >= ROWS * D_INNER) return;
  int row = i / D_INNER, d = i - row * D_INNER;
  int b = row / LL, l = row - b * LL;
  float v = cb[d];
  #pragma unroll
  for (int t = 0; t < D_CONV; ++t) {
    int l2 = l + t - (D_CONV - 1);
    if (l2 >= 0)
      v += xr[((long)(b * LL + l2)) * (2 * D_INNER) + d] * cw[t * D_INNER + d];
  }
  v = v / (1.f + __expf(-v));
  xi[i]  = v;
  xib[i] = f2bf(v);
}

// ================================================================ dt_proj (K=48) + softplus
__global__ __launch_bounds__(256) void k_dt(const float* __restrict__ xdbl,
                                            const float* __restrict__ dtw,
                                            const float* __restrict__ dtb,
                                            float* __restrict__ delta) {
  __shared__ float r[DT_RANK];
  int row = blockIdx.y;
  int d = blockIdx.x * 256 + threadIdx.x;
  if (threadIdx.x < DT_RANK) r[threadIdx.x] = xdbl[(long)row * XDBL_W + threadIdx.x];
  __syncthreads();
  float acc = dtb[d];
  #pragma unroll 8
  for (int k = 0; k < DT_RANK; ++k) acc += r[k] * dtw[k * D_INNER + d];
  float sp = (acc > 20.f) ? acc : logf(1.f + expf(acc));
  delta[(long)row * D_INNER + d] = sp;
}

// ================================================================ selective scan
__global__ __launch_bounds__(256) void k_scan(const float* __restrict__ xi,
                                              const float* __restrict__ delta,
                                              const float* __restrict__ xdbl,
                                              const float* __restrict__ A_log,
                                              const float* __restrict__ Dv,
                                              float* __restrict__ y) {
  __shared__ float sBC[2 * N_STATE];
  const int chunks = D_INNER / 256;
  int b = blockIdx.x / chunks;
  int d = (blockIdx.x % chunks) * 256 + threadIdx.x;
  float A[N_STATE], h[N_STATE];
  #pragma unroll
  for (int n = 0; n < N_STATE; ++n) {
    A[n] = -expf(A_log[(long)d * N_STATE + n]);
    h[n] = 0.f;
  }
  float Dd = Dv[d];
  for (int l = 0; l < LL; ++l) {
    long row = (long)b * LL + l;
    if (threadIdx.x < 2 * N_STATE)
      sBC[threadIdx.x] = xdbl[row * XDBL_W + DT_RANK + threadIdx.x];
    __syncthreads();
    float dl = delta[row * D_INNER + d];
    float u  = xi[row * D_INNER + d];
    float du = dl * u;
    float acc = 0.f;
    #pragma unroll
    for (int n = 0; n < N_STATE; ++n) {
      float dA = expf(dl * A[n]);
      h[n] = dA * h[n] + du * sBC[n];
      acc += h[n] * sBC[N_STATE + n];
    }
    y[row * D_INNER + d] = acc + u * Dd;
    __syncthreads();
  }
}

// ================================================================ gate: yg = bf16(y * silu(res))
__global__ void k_gate(const float* __restrict__ y, const float* __restrict__ xr,
                       __bf16* __restrict__ yg) {
  int i = blockIdx.x * blockDim.x + threadIdx.x;
  if (i >= ROWS * D_INNER) return;
  int row = i / D_INNER, d = i - row * D_INNER;
  float res = xr[(long)row * (2 * D_INNER) + D_INNER + d];
  yg[i] = f2bf(y[i] * (res / (1.f + __expf(-res))));
}

// ================================================================ driver
extern "C" void kernel_launch(void* const* d_in, const int* in_sizes, int n_in,
                              void* d_out, int out_size, void* d_ws, size_t ws_size,
                              hipStream_t stream) {
  const int*   ids   = (const int*)  d_in[0];
  const float* emb   = (const float*)d_in[1];
  const float* normw = (const float*)d_in[2];
  const float* inw   = (const float*)d_in[3];
  const float* convw = (const float*)d_in[4];
  const float* convb = (const float*)d_in[5];
  const float* xpw   = (const float*)d_in[6];
  const float* dtw   = (const float*)d_in[7];
  const float* dtb   = (const float*)d_in[8];
  const float* alog  = (const float*)d_in[9];
  const float* Dp    = (const float*)d_in[10];
  const float* outw  = (const float*)d_in[11];
  const float* normf = (const float*)d_in[12];
  float* out = (float*)d_out;

  // ---- workspace layout (fp32 region then bf16 region) ----
  float* wf    = (float*)d_ws;
  float* x     = wf; wf += (long)ROWS * D_MODEL;
  float* xr    = wf; wf += (long)ROWS * 2 * D_INNER;
  float* xi    = wf; wf += (long)ROWS * D_INNER;
  float* xdbl  = wf; wf += (long)ROWS * XDBL_W;
  float* delta = wf; wf += (long)ROWS * D_INNER;
  float* yb    = wf; wf += (long)ROWS * D_INNER;

  __bf16* wb    = (__bf16*)wf;
  __bf16* embb  = wb; wb += (long)VOCAB * D_MODEL;
  __bf16* inwb  = wb; wb += (long)N_LAYER * 2 * D_INNER * D_MODEL;   // [N=3072][K=768]
  __bf16* xpwb  = wb; wb += (long)N_LAYER * XPW_NPAD * D_INNER;      // [Npad=256][K=1536]
  __bf16* outwb = wb; wb += (long)N_LAYER * D_MODEL * D_INNER;       // [N=768][K=1536]
  __bf16* xnb   = wb; wb += (long)ROWS * D_MODEL;
  __bf16* xib   = wb; wb += (long)ROWS * D_INNER;
  __bf16* ygb   = wb; wb += (long)ROWS * D_INNER;

  // ---- one-time per launch: weights -> bf16, K-contiguous ----
  {
    long n = (long)VOCAB * D_MODEL;
    k_cvt_bf16<<<(unsigned)((n + 255) / 256), 256, 0, stream>>>(emb, embb, n);
  }
  for (int layer = 0; layer < N_LAYER; ++layer) {
    long n1 = (long)2 * D_INNER * D_MODEL;
    k_w_transpose<<<(unsigned)((n1 + 255) / 256), 256, 0, stream>>>(
        inw + (long)layer * D_MODEL * 2 * D_INNER,
        inwb + (long)layer * 2 * D_INNER * D_MODEL, D_MODEL, 2 * D_INNER);
    long n2 = (long)XDBL_W * D_INNER;
    k_w_transpose<<<(unsigned)((n2 + 255) / 256), 256, 0, stream>>>(
        xpw + (long)layer * D_INNER * XDBL_W,
        xpwb + (long)layer * XPW_NPAD * D_INNER, D_INNER, XDBL_W);
    long n3 = (long)D_MODEL * D_INNER;
    k_w_transpose<<<(unsigned)((n3 + 255) / 256), 256, 0, stream>>>(
        outw + (long)layer * D_INNER * D_MODEL,
        outwb + (long)layer * D_MODEL * D_INNER, D_INNER, D_MODEL);
  }

  k_embed<<<(ROWS * D_MODEL + 255) / 256, 256, 0, stream>>>(ids, emb, x);

  for (int layer = 0; layer < N_LAYER; ++layer) {
    k_rmsnorm<<<ROWS, 256, 0, stream>>>(x, normw + layer * D_MODEL, xnb);

    k_gemm_bf16<false><<<dim3(2 * D_INNER / BN, ROWS / BM), 256, 0, stream>>>(
        xnb, inwb + (long)layer * 2 * D_INNER * D_MODEL, xr,
        2 * D_INNER, D_MODEL, 2 * D_INNER);

    k_conv_silu<<<(ROWS * D_INNER + 255) / 256, 256, 0, stream>>>(
        xr, convw + (long)layer * D_CONV * D_INNER, convb + layer * D_INNER, xi, xib);

    k_gemm_bf16<false><<<dim3(1, ROWS / BM), 256, 0, stream>>>(
        xib, xpwb + (long)layer * XPW_NPAD * D_INNER, xdbl,
        XDBL_W, D_INNER, XDBL_W);

    k_dt<<<dim3(D_INNER / 256, ROWS), 256, 0, stream>>>(
        xdbl, dtw + (long)layer * DT_RANK * D_INNER, dtb + layer * D_INNER, delta);

    k_scan<<<BB * (D_INNER / 256), 256, 0, stream>>>(
        xi, delta, xdbl, alog + (long)layer * D_INNER * N_STATE,
        Dp + layer * D_INNER, yb);

    k_gate<<<(ROWS * D_INNER + 255) / 256, 256, 0, stream>>>(yb, xr, ygb);

    k_gemm_bf16<true><<<dim3(D_MODEL / BN, ROWS / BM), 256, 0, stream>>>(
        ygb, outwb + (long)layer * D_MODEL * D_INNER, x,
        D_MODEL, D_INNER, D_MODEL);
  }

  k_rmsnorm<<<ROWS, 256, 0, stream>>>(x, normf, xnb);

  // logits = xn @ emb^T  (dominant GEMM: 4096 x 32000 x 768; emb already [N][K])
  k_gemm_bf16<false><<<dim3(VOCAB / BN, ROWS / BM), 256, 0, stream>>>(
      xnb, embb, out, VOCAB, D_MODEL, VOCAB);
}
